// MultiHeadAttention_82102594831033
// MI455X (gfx1250) — compile-verified, matrix operand order checked by reading, not run on hardware
//
#include <hip/hip_runtime.h>

typedef __attribute__((ext_vector_type(16))) _Float16 v16h;
typedef __attribute__((ext_vector_type(8)))  float    v8f;
typedef __attribute__((ext_vector_type(4)))  int      i32x4;

#define D_MODEL 768
#define N_HEADS 12
#define HEAD_DIM 64
#define BATCH 2
#define SEQ 4096
#define MTOT (BATCH * SEQ)

// ---------------------------------------------------------------------------
// gfx1250 async global->LDS copy (ASYNCcnt-tracked), with portable fallback.
// Probed signature: (int4 __device__*, int4 __shared__*, imm offset, imm cpol)
// ---------------------------------------------------------------------------
#if defined(__AMDGCN__) && __has_builtin(__builtin_amdgcn_global_load_async_to_lds_b128)
#define HAVE_ASYNC_LDS 1
typedef __attribute__((address_space(1))) i32x4* as1_v4i;
typedef __attribute__((address_space(3))) i32x4* as3_v4i;
__device__ __forceinline__ void async_cp16(const _Float16* g, _Float16* l) {
    __builtin_amdgcn_global_load_async_to_lds_b128(
        (as1_v4i)(unsigned long long)g,
        (as3_v4i)(unsigned int)(unsigned long long)l, 0, 0);
}
#else
#define HAVE_ASYNC_LDS 0
__device__ __forceinline__ void async_cp16(const _Float16* g, _Float16* l) {
    *(uint4*)l = *(const uint4*)g;
}
#endif

template <int N>
__device__ __forceinline__ void wait_async() {
#if defined(__AMDGCN__) && __has_builtin(__builtin_amdgcn_s_wait_asynccnt)
    __builtin_amdgcn_s_wait_asynccnt(N);
#elif defined(__AMDGCN__) && HAVE_ASYNC_LDS
    asm volatile("s_wait_asynccnt %0" ::"i"(N) : "memory");
#endif
}

// ---------------------------------------------------------------------------
// WMMA helper: D = A(16x32 f16) * B(32x16 f16) + C(16x16 f32)
// ---------------------------------------------------------------------------
__device__ __forceinline__ v8f wmma_f16(v16h a, v16h b, v8f c) {
    return __builtin_amdgcn_wmma_f32_16x16x32_f16(false, a, false, b,
                                                  (short)0, c, false, false);
}

// A/B fragment where the varying (K) index runs CONTIGUOUSLY in memory.
//   hi=0: e0..7 -> K0..7,  e8..15 -> K16..23
//   hi=1: e0..7 -> K8..15, e8..15 -> K24..31
__device__ __forceinline__ v16h frag_contig(const _Float16* p, int hi) {
    v16h f;
    const _Float16* q0 = p + (hi ? 8 : 0);
    const _Float16* q1 = p + (hi ? 24 : 16);
#pragma unroll
    for (int e = 0; e < 8; ++e) f[e] = q0[e];
#pragma unroll
    for (int e = 0; e < 8; ++e) f[8 + e] = q1[e];
    return f;
}

// ---------------------------------------------------------------------------
// Projection GEMM: Y = X * W^T + bias
// MODE 0: f16 head-major [B, H, L, 64], scaled        (Q, K)
// MODE 2: f16 transposed head-major [B, H, 64, L]     (V)
// MODE 1: f32 row-major [MTOT, 768]                   (final output)
// Block: 256 threads (8 waves) -> 128x64 tile; wave = 32x32 (2x2 WMMA tiles)
// ---------------------------------------------------------------------------
template <bool X_IS_F16, int MODE>
__global__ __launch_bounds__(256) void proj_kernel(
    const float* __restrict__ Xf, const _Float16* __restrict__ Xh,
    const float* __restrict__ W, const float* __restrict__ bias,
    _Float16* __restrict__ out16, float* __restrict__ out32, float scale) {
    const int K = D_MODEL, N = D_MODEL;
    __shared__ __align__(16) _Float16 xs[128][40];
    __shared__ __align__(16) _Float16 wb[64][40];

    const int tid = threadIdx.x;
    const int lane = tid & 31, wid = tid >> 5;
    const int hi = lane >> 4, ln = lane & 15;
    const int m0 = blockIdx.x * 128;
    const int n0 = blockIdx.y * 64;
    const int wm = (wid & 3) * 32;
    const int wn = (wid >> 2) * 32;

    v8f acc[2][2];
#pragma unroll
    for (int i = 0; i < 2; ++i)
#pragma unroll
        for (int j = 0; j < 2; ++j) acc[i][j] = (v8f){0, 0, 0, 0, 0, 0, 0, 0};

    for (int kk = 0; kk < K; kk += 32) {
        __syncthreads();
#pragma unroll
        for (int i = 0; i < 16; ++i) {
            int idx = i * 256 + tid;
            int r = idx >> 5, c = idx & 31;
            size_t g = (size_t)(m0 + r) * K + kk + c;
            float v = X_IS_F16 ? (float)Xh[g] : Xf[g];
            xs[r][c] = (_Float16)v;
        }
#pragma unroll
        for (int i = 0; i < 8; ++i) {
            int idx = i * 256 + tid;
            int r = idx >> 5, c = idx & 31;
            wb[r][c] = (_Float16)W[(size_t)(n0 + r) * K + kk + c];
        }
        __syncthreads();

        v16h a0 = frag_contig(&xs[wm + ln][0], hi);
        v16h a1 = frag_contig(&xs[wm + 16 + ln][0], hi);
        v16h b0 = frag_contig(&wb[wn + ln][0], hi);
        v16h b1 = frag_contig(&wb[wn + 16 + ln][0], hi);
        acc[0][0] = wmma_f16(a0, b0, acc[0][0]);
        acc[0][1] = wmma_f16(a0, b1, acc[0][1]);
        acc[1][0] = wmma_f16(a1, b0, acc[1][0]);
        acc[1][1] = wmma_f16(a1, b1, acc[1][1]);
    }

#pragma unroll
    for (int im = 0; im < 2; ++im) {
#pragma unroll
        for (int in = 0; in < 2; ++in) {
            int col = n0 + wn + in * 16 + ln;
            float bv = bias[col];
#pragma unroll
            for (int r = 0; r < 8; ++r) {
                int row = m0 + wm + im * 16 + r + hi * 8;
                float val = acc[im][in][r] + bv;
                if (MODE == 1) {
                    out32[(size_t)row * N + col] = val;
                } else {
                    val *= scale;
                    int b = row >> 12, l = row & 4095;  // row = b*SEQ + l
                    int h = col >> 6, d = col & 63;     // col = h*64 + d
                    size_t o;
                    if (MODE == 0)
                        o = ((size_t)(b * N_HEADS + h) * SEQ + l) * HEAD_DIM + d;
                    else  // MODE 2: transposed head-major for V
                        o = ((size_t)(b * N_HEADS + h) * HEAD_DIM + d) * SEQ + l;
                    out16[o] = (_Float16)val;
                }
            }
        }
    }
}

// ---------------------------------------------------------------------------
// Causal flash attention. Q/K head-major [B*H, L, 64] f16 (Q pre-scaled),
// V transposed head-major [B*H, 64, L] f16. Block = 128 threads (4 waves),
// 64 query rows per block, 32-key K/V tiles double-buffered via async LDS
// copies (prefetch j+1, s_wait_asynccnt 4 keeps them in flight).
// ---------------------------------------------------------------------------
#define KP 72  // kbuf row pitch (halves) : 144B, 16B-aligned rows
#define VP 40  // vbufT/pbuf row pitch    : 80B,  16B-aligned rows

__global__ __launch_bounds__(128) void attn_kernel(
    const _Float16* __restrict__ Qh, const _Float16* __restrict__ Kh,
    const _Float16* __restrict__ Vt, _Float16* __restrict__ ctx) {
    __shared__ __align__(16) _Float16 kbuf[2][32][KP];   // [buf][key][dim]
    __shared__ __align__(16) _Float16 vbufT[2][64][VP];  // [buf][dim][key]
    __shared__ __align__(16) _Float16 pbuf[4][16][VP];

    const int tid = threadIdx.x;
    const int lane = tid & 31, w = tid >> 5;
    const int hi = lane >> 4, ln = lane & 15;
    const int bh = blockIdx.x;
    const int qblk = blockIdx.y;
    const int b = bh / N_HEADS, h = bh % N_HEADS;
    const size_t base = (size_t)bh * SEQ * HEAD_DIM;  // same extent both layouts
    const int qbase = qblk * 64 + w * 16;

    // per-thread cooperative-copy coordinates (4 x b128 per thread per tile)
    const int kr0 = tid >> 2, kc0 = (tid & 3) * 16;  // K rows 0..31, 2 chunks
    const int vd0 = tid >> 1, vc0 = (tid & 1) * 16;  // V dims 0..63, 2 chunks

    const _Float16* qrow = Qh + base + (size_t)(qbase + ln) * HEAD_DIM;
    v16h qa0 = frag_contig(qrow, hi);
    v16h qa1 = frag_contig(qrow + 32, hi);

    v8f o[4];
#pragma unroll
    for (int g = 0; g < 4; ++g) o[g] = (v8f){0, 0, 0, 0, 0, 0, 0, 0};
    float m[8], l[8];
#pragma unroll
    for (int r = 0; r < 8; ++r) {
        m[r] = -1e30f;
        l[r] = 0.0f;
    }

    const int nkb = qblk * 2 + 2;

    // prefetch tile 0 into buffer 0
    {
        const _Float16* kg = Kh + base + (size_t)kr0 * HEAD_DIM + kc0;
        const _Float16* vg = Vt + base + (size_t)vd0 * SEQ + vc0;
        async_cp16(kg, &kbuf[0][kr0][kc0]);
        async_cp16(kg + 8, &kbuf[0][kr0][kc0 + 8]);
        async_cp16(vg, &vbufT[0][vd0][vc0]);
        async_cp16(vg + 8, &vbufT[0][vd0][vc0 + 8]);
    }

    for (int j = 0; j < nkb; ++j) {
        const int kb = j * 32;
        const int cur = j & 1, nxt = cur ^ 1;
        // prefetch tile j+1 into the other buffer (safe: barrier at loop end
        // guaranteed everyone finished reading it as tile j-1)
        if (j + 1 < nkb) {
            const int kb1 = kb + 32;
            const _Float16* kg = Kh + base + (size_t)(kb1 + kr0) * HEAD_DIM + kc0;
            const _Float16* vg = Vt + base + (size_t)vd0 * SEQ + kb1 + vc0;
            async_cp16(kg, &kbuf[nxt][kr0][kc0]);
            async_cp16(kg + 8, &kbuf[nxt][kr0][kc0 + 8]);
            async_cp16(vg, &vbufT[nxt][vd0][vc0]);
            async_cp16(vg + 8, &vbufT[nxt][vd0][vc0 + 8]);
            wait_async<4>();  // tile j done for this wave; j+1 stays in flight
        } else {
            wait_async<0>();
        }
        __syncthreads();  // tile j visible from all waves

        // S = Q * K^T
        v8f s[2];
#pragma unroll
        for (int g = 0; g < 2; ++g) {
            v8f sc = (v8f){0, 0, 0, 0, 0, 0, 0, 0};
            v16h bk0 = frag_contig(&kbuf[cur][g * 16 + ln][0], hi);
            v16h bk1 = frag_contig(&kbuf[cur][g * 16 + ln][32], hi);
            sc = wmma_f16(qa0, bk0, sc);
            sc = wmma_f16(qa1, bk1, sc);
            s[g] = sc;
        }

        // causal mask only on diagonal blocks (wave-uniform branch)
        if (kb + 31 > qbase) {
            const int kc0m = kb + ln, kc1m = kb + 16 + ln;
#pragma unroll
            for (int r = 0; r < 8; ++r) {
                int qr = qbase + r + hi * 8;
                if (kc0m > qr) s[0][r] = -1e30f;
                if (kc1m > qr) s[1][r] = -1e30f;
            }
        }

        // online softmax (row stats per lane-half)
#pragma unroll
        for (int r = 0; r < 8; ++r) {
            float s0 = s[0][r];
            float s1 = s[1][r];
            float v = fmaxf(s0, s1);
            v = fmaxf(v, __shfl_xor(v, 1));
            v = fmaxf(v, __shfl_xor(v, 2));
            v = fmaxf(v, __shfl_xor(v, 4));
            v = fmaxf(v, __shfl_xor(v, 8));
            float mn = fmaxf(m[r], v);
            float corr = __expf(m[r] - mn);
            float p0 = __expf(s0 - mn);
            float p1 = __expf(s1 - mn);
            float t = p0 + p1;
            t += __shfl_xor(t, 1);
            t += __shfl_xor(t, 2);
            t += __shfl_xor(t, 4);
            t += __shfl_xor(t, 8);
            l[r] = l[r] * corr + t;
            m[r] = mn;
#pragma unroll
            for (int g = 0; g < 4; ++g) o[g][r] *= corr;
            pbuf[w][r + hi * 8][ln] = (_Float16)p0;
            pbuf[w][r + hi * 8][16 + ln] = (_Float16)p1;
        }
        __builtin_amdgcn_wave_barrier();

        // O += P(16x32) * V(32x64) ; B-frag rows come from V^T (contiguous K)
        v16h pa = frag_contig(&pbuf[w][ln][0], hi);
#pragma unroll
        for (int g = 0; g < 4; ++g) {
            v16h vb = frag_contig(&vbufT[cur][g * 16 + ln][0], hi);
            o[g] = wmma_f16(pa, vb, o[g]);
        }
        __syncthreads();  // done reading buf `cur`; next iter may overwrite it
    }

#pragma unroll
    for (int g = 0; g < 4; ++g) {
        int d = g * 16 + ln;
#pragma unroll
        for (int r = 0; r < 8; ++r) {
            int row = qbase + r + hi * 8;
            float val = o[g][r] / l[r];
            ctx[(size_t)(b * SEQ + row) * D_MODEL + h * HEAD_DIM + d] =
                (_Float16)val;
        }
    }
}

// ---------------------------------------------------------------------------
// Launch
// ---------------------------------------------------------------------------
extern "C" void kernel_launch(void* const* d_in, const int* in_sizes, int n_in,
                              void* d_out, int out_size, void* d_ws,
                              size_t ws_size, hipStream_t stream) {
    (void)in_sizes; (void)n_in; (void)out_size; (void)ws_size;

    const float* q  = (const float*)d_in[0];
    const float* k  = (const float*)d_in[1];
    const float* v  = (const float*)d_in[2];
    // d_in[3]: causal mask (recomputed analytically in-kernel)
    const float* wq = (const float*)d_in[4];
    const float* bq = (const float*)d_in[5];
    const float* wk = (const float*)d_in[6];
    const float* bk = (const float*)d_in[7];
    const float* wv = (const float*)d_in[8];
    const float* bv = (const float*)d_in[9];
    const float* wo = (const float*)d_in[10];
    const float* bo = (const float*)d_in[11];
    float* out = (float*)d_out;

    const size_t per = (size_t)BATCH * N_HEADS * SEQ * HEAD_DIM;
    _Float16* qh  = (_Float16*)d_ws;
    _Float16* kh  = qh + per;
    _Float16* vt  = kh + per;
    _Float16* ctx = vt + per;

    dim3 pgrid(MTOT / 128, D_MODEL / 64);  // (64, 12)
    proj_kernel<false, 0><<<pgrid, 256, 0, stream>>>(q, nullptr, wq, bq, qh,
                                                     nullptr, 0.125f);
    proj_kernel<false, 0><<<pgrid, 256, 0, stream>>>(k, nullptr, wk, bk, kh,
                                                     nullptr, 1.0f);
    proj_kernel<false, 2><<<pgrid, 256, 0, stream>>>(v, nullptr, wv, bv, vt,
                                                     nullptr, 1.0f);

    dim3 agrid(BATCH * N_HEADS, SEQ / 64);  // (24, 64)
    attn_kernel<<<agrid, 128, 0, stream>>>(qh, kh, vt, ctx);

    proj_kernel<true, 1><<<pgrid, 256, 0, stream>>>(nullptr, ctx, wo, bo,
                                                    nullptr, out, 1.0f);
}
